// ViTMultiHeadAttention_33397665694315
// MI455X (gfx1250) — compile-verified, matrix operand order checked by reading, not run on hardware
//
#include <hip/hip_runtime.h>
#include <hip/hip_bf16.h>
#include <math.h>

// ---------------------------------------------------------------------------
// Types for CDNA5 WMMA (wave32, 16x16x32 bf16 -> f32)
// ---------------------------------------------------------------------------
typedef __bf16 bf16;
typedef __attribute__((ext_vector_type(8)))  __bf16 v8bf;
typedef __attribute__((ext_vector_type(16))) __bf16 v16bf;
typedef __attribute__((ext_vector_type(8)))  float  v8f;

#define B_   16
#define T_   1029
#define C_   1024
#define H_   16
#define D_   64
#define M_   (B_ * T_)   // 16464 rows (tokens)
#define N1_  (3 * C_)    // 3072 qkv cols
#define TP_  1056        // padded token stride for V^T (covers j=1024 chunk reads, 16B aligned)

__device__ __forceinline__ v16bf frag_cat(v8bf lo, v8bf hi) {
  return __builtin_shufflevector(lo, hi, 0,1,2,3,4,5,6,7,8,9,10,11,12,13,14,15);
}

__device__ __forceinline__ v8f wmma_bf16(v16bf a, v16bf b, v8f c) {
  // D = A(16x32) * B(32x16) + C(16x16 f32)
  return __builtin_amdgcn_wmma_f32_16x16x32_bf16(false, a, false, b, (short)0, c, false, false);
}

__device__ __forceinline__ v8f vzero8() {
  v8f z;
#pragma unroll
  for (int i = 0; i < 8; ++i) z[i] = 0.f;
  return z;
}

// ---------------------------------------------------------------------------
// Shared GEMM mainloop: block tile 128(M) x 128(N), K-step 32, 8 waves.
// Wave w: wm = w>>1 (32-row strip = 2 m-subtiles), wn = w&1 (64-col strip).
// Per K-step each wave: 2 A-frags + 4 B-frags -> 8 WMMAs (each B-frag feeds
// two WMMAs back-to-back). LDS double-buffered with ONE barrier per K-step;
// stage i+1 global loads are issued AFTER the barrier so the pre-barrier wait
// is DScnt-only and the loads overlap stage i's WMMA chain (their loadcnt
// wait lands after the compute, at the cvt/ds_store staging point).
// LDS layout is pre-swizzled into the WMMA fragment layout (ISA 7.12.2):
//   16-bit A 16x32: lane<16 row=lane k={0..7,16..23}; lane>=16 row=lane-16
//   k={8..15,24..31} -> each lane's fragment = two contiguous 16B chunks.
// ---------------------------------------------------------------------------
template <bool ABF>
__device__ __forceinline__ void gemm_mainloop(const void* __restrict__ Aptr,
                                              const float* __restrict__ Bsrc,
                                              int Nld, long m0, long n0b,
                                              v8f acc[2][4]) {
  __shared__ bf16 As[2][8][32][16];   // [stage][m-subtile][lane][16]
  __shared__ bf16 Bs[2][8][32][16];   // [stage][n-subtile][lane][16]
  const int tid  = threadIdx.x;
  const int lane = tid & 31;
  const int w    = tid >> 5;
  const int wm   = w >> 1;
  const int wn   = w & 1;
  // A loader: 128 rows x 4 chunks(8 k) = 512 assignments, 2 per thread
  const int ar = tid >> 2;            // row 0..63 (+64 for second)
  const int ac = tid & 3;             // k-chunk
  const int al = (ar & 15) | ((ac & 1) << 4);
  const int ah = (ac >> 1) * 8;
  // B loader: 128 cols x 4 k-chunks = 512 assignments, 2 per thread
  const int bn  = tid & 127;
  const int bk0 = tid >> 7;
  long arow0 = m0 + ar;      if (arow0 >= M_) arow0 = M_ - 1;
  long arow1 = m0 + ar + 64; if (arow1 >= M_) arow1 = M_ - 1;
#pragma unroll
  for (int i = 0; i < 2; ++i)
#pragma unroll
    for (int s = 0; s < 4; ++s) acc[i][s] = vzero8();

  v8bf ra[2], rb[2];
  auto loadA = [&](int k0) {
#pragma unroll
    for (int j = 0; j < 2; ++j) {
      const long arow = j ? arow1 : arow0;
      if (ABF) {
        ra[j] = *(const v8bf*)((const bf16*)Aptr + arow * (long)C_ + k0 + ac * 8);
      } else {
        const float* ap = (const float*)Aptr + arow * (long)C_ + k0 + ac * 8;
        float4 f0 = *(const float4*)ap;
        float4 f1 = *(const float4*)(ap + 4);
        v8bf av;
        av[0] = (bf16)f0.x; av[1] = (bf16)f0.y; av[2] = (bf16)f0.z; av[3] = (bf16)f0.w;
        av[4] = (bf16)f1.x; av[5] = (bf16)f1.y; av[6] = (bf16)f1.z; av[7] = (bf16)f1.w;
        ra[j] = av;
      }
    }
  };
  auto loadB = [&](int k0) {
#pragma unroll
    for (int c = 0; c < 2; ++c) {
      const int kc = bk0 + 2 * c;
      const float* bp = Bsrc + (long)(k0 + kc * 8) * Nld + n0b + bn;
      v8bf bv;
#pragma unroll
      for (int j = 0; j < 8; ++j) bv[j] = (bf16)bp[(long)j * Nld];
      rb[c] = bv;
    }
    if (k0 + 32 < C_)
      __builtin_prefetch(Bsrc + (long)(k0 + 32) * Nld + n0b + bn, 0, 1);
  };
  auto stage = [&](int st) {
    *(v8bf*)&As[st][(ar >> 4)][al][ah]     = ra[0];
    *(v8bf*)&As[st][(ar >> 4) + 4][al][ah] = ra[1];
#pragma unroll
    for (int c = 0; c < 2; ++c) {
      const int kc = bk0 + 2 * c;
      *(v8bf*)&Bs[st][bn >> 4][(bn & 15) | ((kc & 1) << 4)][(kc >> 1) * 8] = rb[c];
    }
  };

  loadA(0); loadB(0); stage(0);
  const int NK = C_ / 32;
  for (int i = 0; i < NK; ++i) {
    const int cur = i & 1;
    __syncthreads();   // DScnt-only wait: stage 'cur' stores visible everywhere
    // Issue next stage's global loads AFTER the barrier: they fly during the
    // WMMA chain; their wait lands at the cvt/ds_store staging below.
    if (i + 1 < NK) { loadA((i + 1) * 32); loadB((i + 1) * 32); }
    v16bf af0 = frag_cat(*(const v8bf*)&As[cur][2 * wm][lane][0],
                         *(const v8bf*)&As[cur][2 * wm][lane][8]);
    v16bf af1 = frag_cat(*(const v8bf*)&As[cur][2 * wm + 1][lane][0],
                         *(const v8bf*)&As[cur][2 * wm + 1][lane][8]);
#pragma unroll
    for (int s = 0; s < 4; ++s) {
      v16bf bfr = frag_cat(*(const v8bf*)&Bs[cur][wn * 4 + s][lane][0],
                           *(const v8bf*)&Bs[cur][wn * 4 + s][lane][8]);
      acc[0][s] = wmma_bf16(af0, bfr, acc[0][s]);
      acc[1][s] = wmma_bf16(af1, bfr, acc[1][s]);
    }
    if (i + 1 < NK) stage((i + 1) & 1);
  }
}

// ---------------------------------------------------------------------------
// Kernel 1: qkv = x @ Wqkv + bqkv, fused RoPE on q/k, writes bf16 workspace:
//   Q,K : [B,H,T,64]   (row = token, contiguous along d -> b128 frag loads)
//   V^T : [B,H,64,TP_] (row = d,     contiguous along t -> b128 frag loads)
// Each wave's 64-col strip is exactly one head, so RoPE partner d+32 sits in
// accumulator tile (s+2) of the same lane/register.
// ---------------------------------------------------------------------------
__global__ __launch_bounds__(256) void qkv_rope_kernel(
    const float* __restrict__ x, const float* __restrict__ Wqkv,
    const float* __restrict__ bqkv, const float* __restrict__ cosb,
    const float* __restrict__ sinb, const int* __restrict__ npfx_p,
    bf16* __restrict__ qws, bf16* __restrict__ kws, bf16* __restrict__ vtws) {
  v8f acc[2][4];
  const long m0  = (long)blockIdx.x * 128;
  const long n0b = (long)blockIdx.y * 128;
  gemm_mainloop<false>(x, Wqkv, N1_, m0, n0b, acc);

  const int tid = threadIdx.x, lane = tid & 31, w = tid >> 5;
  const int wm = w >> 1, wn = w & 1;
  const int l16 = lane & 15, lh = lane >> 4;
  const long n0 = n0b + wn * 64;
  const int seg = (int)(n0 / C_);            // 0=q 1=k 2=v
  const int h   = (int)((n0 % C_) / D_);
  const int npfx = *npfx_p;

#pragma unroll
  for (int s = 0; s < 4; ++s) {
    const float bb = bqkv[n0 + s * 16 + l16];
#pragma unroll
    for (int ms = 0; ms < 2; ++ms)
#pragma unroll
      for (int r = 0; r < 8; ++r) acc[ms][s][r] += bb;
  }

#pragma unroll
  for (int ms = 0; ms < 2; ++ms) {
#pragma unroll
    for (int r = 0; r < 8; ++r) {
      const long g = m0 + wm * 32 + ms * 16 + r + 8 * lh;  // C-layout row
      if (g >= M_) continue;
      const int b = (int)(g / T_), t = (int)(g % T_);
      float v0 = acc[ms][0][r], v1 = acc[ms][1][r];
      float v2 = acc[ms][2][r], v3 = acc[ms][3][r];
      if (seg < 2 && t >= npfx) {
        const int pos = t - npfx;
        const float* cr = cosb + (long)pos * D_;
        const float* sr = sinb + (long)pos * D_;
        const int d0 = l16, d1 = 16 + l16;
        const float n0v = v0 * cr[d0]      - v2 * sr[d0];
        const float n1v = v1 * cr[d1]      - v3 * sr[d1];
        const float n2v = v2 * cr[d0 + 32] + v0 * sr[d0 + 32];
        const float n3v = v3 * cr[d1 + 32] + v1 * sr[d1 + 32];
        v0 = n0v; v1 = n1v; v2 = n2v; v3 = n3v;
      }
      const long bh = (long)(b * H_ + h);
      if (seg == 0) {
        bf16* dst = qws + (bh * T_ + t) * D_;
        dst[l16] = (bf16)v0; dst[16 + l16] = (bf16)v1;
        dst[32 + l16] = (bf16)v2; dst[48 + l16] = (bf16)v3;
      } else if (seg == 1) {
        bf16* dst = kws + (bh * T_ + t) * D_;
        dst[l16] = (bf16)v0; dst[16 + l16] = (bf16)v1;
        dst[32 + l16] = (bf16)v2; dst[48 + l16] = (bf16)v3;
      } else {
        bf16* dst = vtws + bh * D_ * TP_ + t;
        dst[(long)(l16)      * TP_] = (bf16)v0;
        dst[(long)(16 + l16) * TP_] = (bf16)v1;
        dst[(long)(32 + l16) * TP_] = (bf16)v2;
        dst[(long)(48 + l16) * TP_] = (bf16)v3;
      }
    }
  }
}

// ---------------------------------------------------------------------------
// Kernel 2: flash attention per (b,h). One 16-query tile per wave, 32 keys
// per step (= WMMA K). Online softmax with 16-lane shuffle reductions; P is
// re-laid-out C->A fragment through per-wave LDS (in-order DS, no barrier).
// K/V fragment loads come straight from the L2-resident bf16 workspace.
// ---------------------------------------------------------------------------
__global__ __launch_bounds__(256) void attn_kernel(
    const bf16* __restrict__ qws, const bf16* __restrict__ kws,
    const bf16* __restrict__ vtws, bf16* __restrict__ yws) {
  __shared__ bf16 Pl[8][16][32];
  const int tid = threadIdx.x, lane = tid & 31, w = tid >> 5;
  const int l16 = lane & 15, lh = lane >> 4, kb = lh * 8;
  const int bh = blockIdx.x, b = bh >> 4, h = bh & 15;
  const int q0 = blockIdx.y * 128 + w * 16;
  if (q0 >= T_) return;

  const bf16* Q  = qws  + (long)bh * T_ * D_;
  const bf16* Kb = kws  + (long)bh * T_ * D_;
  const bf16* Vt = vtws + (long)bh * D_ * TP_;

  int qrow = q0 + l16; if (qrow > T_ - 1) qrow = T_ - 1;
  const bf16* qr = Q + (long)qrow * D_;
  const v16bf aq0 = frag_cat(*(const v8bf*)(qr + kb),      *(const v8bf*)(qr + kb + 16));
  const v16bf aq1 = frag_cat(*(const v8bf*)(qr + 32 + kb), *(const v8bf*)(qr + 32 + kb + 16));

  v8f o[4];
  float mr[8], lr[8];
#pragma unroll
  for (int s = 0; s < 4; ++s) o[s] = vzero8();
#pragma unroll
  for (int r = 0; r < 8; ++r) { mr[r] = -3.0e38f; lr[r] = 0.f; }

  for (int j = 0; j < T_; j += 32) {
    int kt0 = j + l16;      const bool ok0 = kt0 < T_; if (!ok0) kt0 = T_ - 1;
    int kt1 = j + 16 + l16; const bool ok1 = kt1 < T_; if (!ok1) kt1 = T_ - 1;
    const bf16* kr0 = Kb + (long)kt0 * D_;
    const bf16* kr1 = Kb + (long)kt1 * D_;
    // S = Q K^T over d=64 (two K=32 WMMAs per 16-key tile)
    v8f s0 = wmma_bf16(aq0, frag_cat(*(const v8bf*)(kr0 + kb), *(const v8bf*)(kr0 + kb + 16)), vzero8());
    s0 = wmma_bf16(aq1, frag_cat(*(const v8bf*)(kr0 + 32 + kb), *(const v8bf*)(kr0 + 32 + kb + 16)), s0);
    v8f s1 = wmma_bf16(aq0, frag_cat(*(const v8bf*)(kr1 + kb), *(const v8bf*)(kr1 + kb + 16)), vzero8());
    s1 = wmma_bf16(aq1, frag_cat(*(const v8bf*)(kr1 + 32 + kb), *(const v8bf*)(kr1 + 32 + kb + 16)), s1);

#pragma unroll
    for (int r = 0; r < 8; ++r) {   // row = r + 8*lh lives across one 16-lane half
      float a0 = ok0 ? s0[r] * 0.125f : -3.0e38f;
      float a1 = ok1 ? s1[r] * 0.125f : -3.0e38f;
      float tm = fmaxf(a0, a1);
#pragma unroll
      for (int msk = 1; msk < 16; msk <<= 1) tm = fmaxf(tm, __shfl_xor(tm, msk, 32));
      const float mn    = fmaxf(mr[r], tm);
      const float alpha = __expf(mr[r] - mn);
      mr[r] = mn;
      const float p0 = __expf(a0 - mn);
      const float p1 = __expf(a1 - mn);
      float ps = p0 + p1;
#pragma unroll
      for (int msk = 1; msk < 16; msk <<= 1) ps += __shfl_xor(ps, msk, 32);
      lr[r] = lr[r] * alpha + ps;
#pragma unroll
      for (int s = 0; s < 4; ++s) o[s][r] *= alpha;
      const int row = r + 8 * lh;
      Pl[w][row][l16]      = (bf16)p0;
      Pl[w][row][16 + l16] = (bf16)p1;
    }
    // P (C-layout) -> A-fragment via wave-private LDS (DS ops are in-order)
    v16bf ap = frag_cat(*(const v8bf*)&Pl[w][l16][kb], *(const v8bf*)&Pl[w][l16][kb + 16]);
#pragma unroll
    for (int s = 0; s < 4; ++s) {
      const bf16* vp = Vt + (long)(s * 16 + l16) * TP_ + j;  // V^T: contiguous tokens
      v16bf bv = frag_cat(*(const v8bf*)(vp + kb), *(const v8bf*)(vp + kb + 16));
      o[s] = wmma_bf16(ap, bv, o[s]);
    }
  }

#pragma unroll
  for (int r = 0; r < 8; ++r) {
    const int t = q0 + r + 8 * lh;
    if (t >= T_) continue;
    const float inv = 1.0f / lr[r];
    bf16* yp = yws + ((long)(b * T_ + t) * C_) + h * D_;
#pragma unroll
    for (int s = 0; s < 4; ++s) yp[s * 16 + l16] = (bf16)(o[s][r] * inv);
  }
}

// ---------------------------------------------------------------------------
// Kernel 3: out = y @ Wout + bout (A already bf16 in workspace)
// ---------------------------------------------------------------------------
__global__ __launch_bounds__(256) void out_proj_kernel(
    const bf16* __restrict__ yws, const float* __restrict__ Wout,
    const float* __restrict__ bout, float* __restrict__ out) {
  v8f acc[2][4];
  const long m0  = (long)blockIdx.x * 128;
  const long n0b = (long)blockIdx.y * 128;
  gemm_mainloop<true>(yws, Wout, C_, m0, n0b, acc);

  const int tid = threadIdx.x, lane = tid & 31, w = tid >> 5;
  const int wm = w >> 1, wn = w & 1;
  const int l16 = lane & 15, lh = lane >> 4;
  const long n0 = n0b + wn * 64;
#pragma unroll
  for (int s = 0; s < 4; ++s) {
    const float bb = bout[n0 + s * 16 + l16];
#pragma unroll
    for (int ms = 0; ms < 2; ++ms)
#pragma unroll
      for (int r = 0; r < 8; ++r) {
        const long g = m0 + wm * 32 + ms * 16 + r + 8 * lh;
        if (g < M_) out[g * C_ + n0 + s * 16 + l16] = acc[ms][s][r] + bb;
      }
  }
}

// ---------------------------------------------------------------------------
extern "C" void kernel_launch(void* const* d_in, const int* in_sizes, int n_in,
                              void* d_out, int out_size, void* d_ws, size_t ws_size,
                              hipStream_t stream) {
  (void)in_sizes; (void)n_in; (void)out_size; (void)ws_size;
  const float* x    = (const float*)d_in[0];
  const float* Wqkv = (const float*)d_in[1];
  const float* bqkv = (const float*)d_in[2];
  const float* Wout = (const float*)d_in[3];
  const float* bout = (const float*)d_in[4];
  const float* cosb = (const float*)d_in[5];
  const float* sinb = (const float*)d_in[6];
  const int*   npfx = (const int*)d_in[7];

  bf16* qws  = (bf16*)d_ws;
  bf16* kws  = qws  + (size_t)B_ * H_ * T_ * D_;
  bf16* vtws = kws  + (size_t)B_ * H_ * T_ * D_;
  bf16* yws  = vtws + (size_t)B_ * H_ * D_ * TP_;

  dim3 g1((M_ + 127) / 128, N1_ / 128);
  hipLaunchKernelGGL(qkv_rope_kernel, g1, dim3(256), 0, stream,
                     x, Wqkv, bqkv, cosb, sinb, npfx, qws, kws, vtws);

  dim3 g2(B_ * H_, (T_ + 127) / 128);
  hipLaunchKernelGGL(attn_kernel, g2, dim3(256), 0, stream, qws, kws, vtws, yws);

  dim3 g3((M_ + 127) / 128, C_ / 128);
  hipLaunchKernelGGL(out_proj_kernel, g3, dim3(256), 0, stream,
                     yws, Wout, bout, (float*)d_out);
}
